// _MSTGCNBlock_86655260164725
// MI455X (gfx1250) — matrix operand order repaired and checked by reading, not compiled, not run on hardware
//
#include <hip/hip_runtime.h>
#include <hip/hip_bf16.h>

// ---------------------------------------------------------------------------
// MSTGCN block, fully fused for MI455X (gfx1250, wave32, WMMA bf16 16x16x32).
// Dominant GEMM: rhs[b,k] = cheb[k]^T @ x[b]  (2048x2048 @ 2048x256, 96x)
// Tiles are staged with GLOBAL_LOAD_ASYNC_TO_LDS_B128 (ASYNCcnt) into
// double-buffered LDS so DMA overlaps WMMA. All downstream ops (Theta
// contraction, 3-tap time conv, residual, LN) stay on-chip.
// ---------------------------------------------------------------------------

typedef __attribute__((ext_vector_type(16))) __bf16 v16bf;
typedef __attribute__((ext_vector_type(8)))  float  v8f;

#define B_DIM 32
#define N_DIM 2048
#define F_INC 16
#define T_DIM 16
#define C_DIM 256   // F_IN * T
#define K_CH  3
#define OCH   64    // CHEB_F == TIME_F

// ---- LDS layout (bytes). Padded strides (72 elems) keep b128 frags 16B
// aligned while spreading banks.
//   consts:               0 .. 5120
//   ping-pong tiles:   5120 .. 106496   (2 x (Xt 36864 + chebA 13824))
//   rhsT (after loop): 5120 .. 78848    (overlaps dead ping-pong bufs)
//   spatialT:        106496 .. 189440   (zeroed up front, outside bufs)
#define CONST_OFF 0
#define PBUF_SZ   50688
#define PBUF_OFF(buf) (5120 + (buf) * PBUF_SZ)
#define XT_BYTES  36864                   // Xt   [256][72] bf16
#define R_OFF     5120                    // rhsT [32][16][72] bf16 = 73728 B
#define S_OFF     106496                  // spatT[32][18][72] bf16 = 82944 B
#define SMEM_BYTES (S_OFF + 82944)        // 189440 B (< 320KB WGP)
#define XT_S 72
#define CB_S 72
#define RT_S 72
#define SP_S 72

// workspace offsets (bytes)
#define WS_XT    0                        // 32*256*2048*2 = 33554432
#define WS_CHEB  33554432                 // 3*2048*2048*2 = 25165824
#define WS_THETA 58720256                 // 64*64*2       = 8192
#define WS_WT    58728448                 // 3*64*64*2     = 24576

static __device__ inline v16bf lds_frag(const char* sm, int e0, int e1) {
  union { v16bf v; uint4 u[2]; } r;
  r.u[0] = *(const uint4*)(sm + 2 * e0);
  r.u[1] = *(const uint4*)(sm + 2 * e1);
  return r.v;
}
static __device__ inline v16bf glb_frag(const __hip_bfloat16* p, int e0, int e1) {
  union { v16bf v; uint4 u[2]; } r;
  r.u[0] = *(const uint4*)(p + e0);
  r.u[1] = *(const uint4*)(p + e1);
  return r.v;
}
static __device__ inline v8f wmma_bf16(v16bf a, v16bf b, v8f c) {
  return __builtin_amdgcn_wmma_f32_16x16x32_bf16(false, a, false, b, (short)0, c,
                                                 false, false);
}
// Async DMA: 32 lanes x 16B -> LDS, tracked by ASYNCcnt (bypasses VGPRs).
static __device__ inline void async_ld_b128(uint32_t lds_byte, uint32_t voff,
                                            unsigned long long base) {
  asm volatile("global_load_async_to_lds_b128 %0, %1, %2"
               :: "v"(lds_byte), "v"(voff), "s"(base)
               : "memory");
}

// ---------------------------------------------------------------------------
// Kernel 1: pack fp32 inputs into bf16 WMMA-friendly layouts.
//   xT[b][c][m]    = x[b][m][f][t],  c = f*16 + t   (B-operand, K=m contiguous)
//   chebT[k][n][m] = cheb[k][m][n]                  (A-operand, K=m contiguous)
//   ThetaA[o][kf]  = Theta[k][f][o], kf = k*16+f, padded 48..63 with 0
//   WtA[d][o][c]   = time_w[o][c][0][d]
// ---------------------------------------------------------------------------
__global__ void mstgcn_prep(const float* __restrict__ x,
                            const float* __restrict__ cheb,
                            const float* __restrict__ Theta,
                            const float* __restrict__ time_w,
                            __hip_bfloat16* __restrict__ xT,
                            __hip_bfloat16* __restrict__ chebT,
                            __hip_bfloat16* __restrict__ ThetaA,
                            __hip_bfloat16* __restrict__ WtA) {
  const long long i0 = (long long)blockIdx.x * blockDim.x + threadIdx.x;
  const long long stride = (long long)gridDim.x * blockDim.x;

  const long long T1 = (long long)B_DIM * C_DIM * N_DIM;     // 16.8M
  for (long long j = i0; j < T1; j += stride) {
    int b = (int)(j / ((long long)C_DIM * N_DIM));
    int rem = (int)(j % ((long long)C_DIM * N_DIM));
    int c = rem / N_DIM, m = rem % N_DIM;
    int f = c >> 4, t = c & 15;
    xT[j] = __float2bfloat16(x[(((long long)(b * N_DIM + m)) * F_INC + f) * T_DIM + t]);
  }
  const long long T2 = (long long)K_CH * N_DIM * N_DIM;      // 12.6M
  for (long long j = i0; j < T2; j += stride) {
    int k = (int)(j / ((long long)N_DIM * N_DIM));
    int rem = (int)(j % ((long long)N_DIM * N_DIM));
    int n = rem / N_DIM, m = rem % N_DIM;
    chebT[j] = __float2bfloat16(cheb[((long long)k * N_DIM + m) * N_DIM + n]);
  }
  for (long long j = i0; j < OCH * 64; j += stride) {
    int o = (int)(j / 64), kf = (int)(j % 64);
    int k = kf >> 4, f = kf & 15;
    float v = (kf < 48) ? Theta[(k * F_INC + f) * OCH + o] : 0.f;
    ThetaA[j] = __float2bfloat16(v);
  }
  for (long long j = i0; j < 3 * OCH * OCH; j += stride) {
    int d = (int)(j / (OCH * OCH));
    int rem = (int)(j % (OCH * OCH));
    int o = rem / OCH, c = rem % OCH;
    WtA[j] = __float2bfloat16(time_w[o * (OCH * 3) + c * 3 + d]);
  }
}

// ---------------------------------------------------------------------------
// Kernel 2: fused main kernel. One WG (256 thr = 8 wave32) per (b, 32-row n
// block). Produces final LN output for its rows.
// ---------------------------------------------------------------------------
__global__ __launch_bounds__(256, 1) void mstgcn_main(
    const float* __restrict__ x,
    const __hip_bfloat16* __restrict__ xT,
    const __hip_bfloat16* __restrict__ chebT,
    const __hip_bfloat16* __restrict__ ThetaA,
    const __hip_bfloat16* __restrict__ WtA,
    const float* __restrict__ time_b,
    const float* __restrict__ res_w,
    const float* __restrict__ res_b,
    const float* __restrict__ ln_g,
    const float* __restrict__ ln_b,
    float* __restrict__ out) {
  extern __shared__ char smem[];
  const int tid = threadIdx.x;
  const int wave = tid >> 5, lane = tid & 31;
  const int lane15 = lane & 15, khalf = lane >> 4;
  const int b = blockIdx.y;
  const int n0 = blockIdx.x * 32;

  float* res_w_s  = (float*)(smem + CONST_OFF);   // [64][16]
  float* res_b_s  = res_w_s + 1024;
  float* time_b_s = res_b_s + 64;
  float* ln_g_s   = time_b_s + 64;
  float* ln_b_s   = ln_g_s + 64;

  // ---- phase 0: consts into LDS; zero spatialT (t and c padding stay 0)
  for (int i = tid; i < 1024; i += 256) res_w_s[i] = res_w[i];
  if (tid < 64) {
    res_b_s[tid]  = res_b[tid];
    time_b_s[tid] = time_b[tid];
    ln_g_s[tid]   = ln_g[tid];
    ln_b_s[tid]   = ln_b[tid];
  }
  {
    uint32_t* spz = (uint32_t*)(smem + S_OFF);
    for (int i = tid; i < (32 * 18 * SP_S * 2) / 4; i += 256) spz[i] = 0u;
  }

  // ---- phase 1: rhs_k(32n x 256c) = sum_m chebT[k][n][m] * xT[b][c][m]
  const int ni = wave & 1;   // n-tile 0/1 (16 rows each)
  const int cg = wave >> 1;  // c-group 0..3 (64 cols each)
  const v8f zero8 = {0.f, 0.f, 0.f, 0.f, 0.f, 0.f, 0.f, 0.f};
  v8f acc[K_CH][4];
#pragma unroll
  for (int k = 0; k < K_CH; ++k)
#pragma unroll
    for (int ct = 0; ct < 4; ++ct) acc[k][ct] = zero8;

  const __hip_bfloat16* xTb = xT + (size_t)b * C_DIM * N_DIM;
  const uint32_t lds32 = (uint32_t)(size_t)(&smem[0]);

  // Per-lane invariant parts of the async-DMA addresses (8 Xt + 3 cheb
  // b128 instructions per wave per 64-deep m tile).
  uint32_t xt_lds[8], xt_gv[8];
#pragma unroll
  for (int i = 0; i < 8; ++i) {
    int chunk = tid + i * 256;                 // 2048 chunks of 8 bf16
    int c = chunk >> 3, m8 = (chunk & 7) * 8;
    xt_lds[i] = 2u * (uint32_t)(c * XT_S + m8);
    xt_gv[i]  = 2u * (uint32_t)(c * N_DIM + m8);
  }
  uint32_t cb_lds[3], cb_gv[3];
#pragma unroll
  for (int j = 0; j < 3; ++j) {
    int chunk = tid + j * 256;                 // 768 chunks of 8 bf16
    int k = chunk >> 8, r = chunk & 255;
    int n = r >> 3, m8 = (r & 7) * 8;
    cb_lds[j] = 2u * (uint32_t)((k * 32 + n) * CB_S + m8);
    cb_gv[j]  = 2u * (uint32_t)((k * N_DIM + n0 + n) * N_DIM + m8);
  }

#define ISSUE_TILE(tile)                                                      \
  do {                                                                        \
    int _buf = (tile) & 1;                                                    \
    uint32_t _pxt = lds32 + (uint32_t)PBUF_OFF(_buf);                         \
    uint32_t _pcb = _pxt + XT_BYTES;                                          \
    uint32_t _g = 2u * (uint32_t)((tile) * 64);                               \
    _Pragma("unroll") for (int _i = 0; _i < 8; ++_i)                          \
        async_ld_b128(_pxt + xt_lds[_i], xt_gv[_i] + _g,                      \
                      (unsigned long long)(size_t)xTb);                       \
    _Pragma("unroll") for (int _j = 0; _j < 3; ++_j)                          \
        async_ld_b128(_pcb + cb_lds[_j], cb_gv[_j] + _g,                      \
                      (unsigned long long)(size_t)chebT);                     \
  } while (0)

  ISSUE_TILE(0);

  for (int it = 0; it < N_DIM / 64; ++it) {
    if (it + 1 < N_DIM / 64) {
      ISSUE_TILE(it + 1);  // prefetch next tile into the other buffer
      // current tile's 11 ops are the oldest; keep next tile's in flight
      asm volatile("s_wait_asynccnt 0xb" ::: "memory");
    } else {
      asm volatile("s_wait_asynccnt 0x0" ::: "memory");
    }
    __syncthreads();

    const char* pxt = smem + PBUF_OFF(it & 1);
    const char* pcb = pxt + XT_BYTES;
#pragma unroll
    for (int kk = 0; kk < 64; kk += 32) {
      const int row = ni * 16 + lane15;
      const int mo = kk + khalf * 8;
      v16bf a0 = lds_frag(pcb, (0 * 32 + row) * CB_S + mo,
                          (0 * 32 + row) * CB_S + mo + 16);
      v16bf a1 = lds_frag(pcb, (1 * 32 + row) * CB_S + mo,
                          (1 * 32 + row) * CB_S + mo + 16);
      v16bf a2 = lds_frag(pcb, (2 * 32 + row) * CB_S + mo,
                          (2 * 32 + row) * CB_S + mo + 16);
#pragma unroll
      for (int ct = 0; ct < 4; ++ct) {
        int col = cg * 64 + ct * 16 + lane15;
        int bo = col * XT_S + kk + khalf * 8;
        v16bf bb = lds_frag(pxt, bo, bo + 16);
        acc[0][ct] = wmma_bf16(a0, bb, acc[0][ct]);
        acc[1][ct] = wmma_bf16(a1, bb, acc[1][ct]);
        acc[2][ct] = wmma_bf16(a2, bb, acc[2][ct]);
      }
    }
    __syncthreads();  // buffer (it&1) free for iteration it+2's DMA
  }

  // ---- phase a: scatter accumulators -> rhsT[n][t][kf] (bf16, kf padded 64)
  {
    const int t = lane15, rhalf = khalf * 8;
#pragma unroll
    for (int k = 0; k < K_CH; ++k)
#pragma unroll
      for (int ct = 0; ct < 4; ++ct) {
        int kf = k * 16 + cg * 4 + ct;
#pragma unroll
        for (int r = 0; r < 8; ++r) {
          int nr = ni * 16 + r + rhalf;
          *(__hip_bfloat16*)(smem + R_OFF + 2 * ((nr * 16 + t) * RT_S + kf)) =
              __float2bfloat16(acc[k][ct][r]);
        }
      }
    for (int i = tid; i < 32 * 16 * 24; i += 256) {  // zero kf pad 48..71
      int nt = i / 24, j = i % 24;
      *(__hip_bfloat16*)(smem + R_OFF + 2 * (nt * RT_S + 48 + j)) =
          __float2bfloat16(0.f);
    }
  }
  __syncthreads();

  // ---- phase b: spatial[o, (n,t)] = relu(ThetaA(64x64) @ rhsT) via WMMA
  v8f acc2[4][4];  // [pt][ot], wave owns n rows wave*4 .. wave*4+3
#pragma unroll
  for (int pt = 0; pt < 4; ++pt)
#pragma unroll
    for (int ot = 0; ot < 4; ++ot) acc2[pt][ot] = zero8;

#pragma unroll
  for (int pt = 0; pt < 4; ++pt) {
    int nl = wave * 4 + pt;
    int be = (nl * 16 + lane15) * RT_S + khalf * 8;
    v16bf b0 = lds_frag(smem + R_OFF, be, be + 16);           // kf 0..31
    v16bf b1 = lds_frag(smem + R_OFF, be + 32, be + 48);      // kf 32..63
#pragma unroll
    for (int ot = 0; ot < 4; ++ot) {
      int ar = (ot * 16 + lane15) * 64 + khalf * 8;
      v16bf ta0 = glb_frag(ThetaA, ar, ar + 16);
      v16bf ta1 = glb_frag(ThetaA, ar + 32, ar + 32 + 16);
      acc2[pt][ot] = wmma_bf16(ta0, b0, acc2[pt][ot]);
      acc2[pt][ot] = wmma_bf16(ta1, b1, acc2[pt][ot]);
    }
  }
  {  // relu + store spatialT[n][1+t][o] (t-padded for the 3-tap conv)
    const int t = lane15, rhalf = khalf * 8;
#pragma unroll
    for (int pt = 0; pt < 4; ++pt) {
      int nl = wave * 4 + pt;
#pragma unroll
      for (int ot = 0; ot < 4; ++ot)
#pragma unroll
        for (int r = 0; r < 8; ++r) {
          int o = ot * 16 + r + rhalf;
          float v = fmaxf(acc2[pt][ot][r], 0.f);
          *(__hip_bfloat16*)(smem + S_OFF +
                             2 * ((nl * 18 + 1 + t) * SP_S + o)) =
              __float2bfloat16(v);
        }
    }
  }
  __syncthreads();

  // ---- phase c: time conv as 3 shifted GEMMs: out += WtA[d] @ spatialT(t+d)
  v8f acc3[4][4];
#pragma unroll
  for (int pt = 0; pt < 4; ++pt)
#pragma unroll
    for (int ot = 0; ot < 4; ++ot) acc3[pt][ot] = zero8;

#pragma unroll
  for (int pt = 0; pt < 4; ++pt) {
    int nl = wave * 4 + pt;
#pragma unroll
    for (int d = 0; d < 3; ++d) {
      int bb = (nl * 18 + lane15 + d) * SP_S + khalf * 8;
      v16bf b0 = lds_frag(smem + S_OFF, bb, bb + 16);         // c 0..31
      v16bf b1 = lds_frag(smem + S_OFF, bb + 32, bb + 48);    // c 32..63
#pragma unroll
      for (int ot = 0; ot < 4; ++ot) {
        int ar = (d * 64 + ot * 16 + lane15) * 64 + khalf * 8;
        v16bf w0 = glb_frag(WtA, ar, ar + 16);
        v16bf w1 = glb_frag(WtA, ar + 32, ar + 32 + 16);
        acc3[pt][ot] = wmma_bf16(w0, b0, acc3[pt][ot]);
        acc3[pt][ot] = wmma_bf16(w1, b1, acc3[pt][ot]);
      }
    }
  }

  // ---- phase d: +bias, +1x1 residual, relu, LayerNorm over 64 ch, store
  {
    const int t = lane15, rhalf = khalf * 8;
#pragma unroll
    for (int pt = 0; pt < 4; ++pt) {
      int nl = wave * 4 + pt;
      int n = n0 + nl;
      float xv[16];
#pragma unroll
      for (int f = 0; f < F_INC; ++f)
        xv[f] = x[(((size_t)(b * N_DIM + n)) * F_INC + f) * T_DIM + t];
      float s = 0.f, sq = 0.f;
#pragma unroll
      for (int ot = 0; ot < 4; ++ot)
#pragma unroll
        for (int r = 0; r < 8; ++r) {
          int o = ot * 16 + r + rhalf;
          float res = res_b_s[o];
#pragma unroll
          for (int f = 0; f < F_INC; ++f)
            res = fmaf(res_w_s[o * 16 + f], xv[f], res);
          float v = acc3[pt][ot][r] + time_b_s[o] + res;
          v = fmaxf(v, 0.f);
          acc3[pt][ot][r] = v;
          s += v;
          sq += v * v;
        }
      // lanes L and L^16 hold complementary channel rows of the same (n,t)
      s  += __shfl_xor(s, 16, 32);
      sq += __shfl_xor(sq, 16, 32);
      float mean = s * (1.f / 64.f);
      float var  = sq * (1.f / 64.f) - mean * mean;
      float rstd = rsqrtf(var + 1e-5f);
#pragma unroll
      for (int ot = 0; ot < 4; ++ot)
#pragma unroll
        for (int r = 0; r < 8; ++r) {
          int o = ot * 16 + r + rhalf;
          out[(((size_t)(b * N_DIM + n)) * OCH + o) * T_DIM + t] =
              (acc3[pt][ot][r] - mean) * rstd * ln_g_s[o] + ln_b_s[o];
        }
    }
  }
}

extern "C" void kernel_launch(void* const* d_in, const int* in_sizes, int n_in,
                              void* d_out, int out_size, void* d_ws,
                              size_t ws_size, hipStream_t stream) {
  const float* x      = (const float*)d_in[0];
  const float* cheb   = (const float*)d_in[1];
  const float* Theta  = (const float*)d_in[2];
  const float* time_w = (const float*)d_in[3];
  const float* time_b = (const float*)d_in[4];
  const float* res_w  = (const float*)d_in[5];
  const float* res_b  = (const float*)d_in[6];
  const float* ln_g   = (const float*)d_in[7];
  const float* ln_b   = (const float*)d_in[8];

  char* ws = (char*)d_ws;
  __hip_bfloat16* xT     = (__hip_bfloat16*)(ws + WS_XT);
  __hip_bfloat16* chebT  = (__hip_bfloat16*)(ws + WS_CHEB);
  __hip_bfloat16* ThetaA = (__hip_bfloat16*)(ws + WS_THETA);
  __hip_bfloat16* WtA    = (__hip_bfloat16*)(ws + WS_WT);

  hipLaunchKernelGGL(mstgcn_prep, dim3(2048), dim3(256), 0, stream, x, cheb,
                     Theta, time_w, xT, chebT, ThetaA, WtA);

  (void)hipFuncSetAttribute((const void*)mstgcn_main,
                            hipFuncAttributeMaxDynamicSharedMemorySize,
                            SMEM_BYTES);
  dim3 grid(N_DIM / 32, B_DIM);
  hipLaunchKernelGGL(mstgcn_main, grid, dim3(256), SMEM_BYTES, stream, x, xT,
                     chebT, ThetaA, WtA, time_b, res_w, res_b, ln_g, ln_b,
                     (float*)d_out);
}
